// LorentzMoE_3891240370246
// MI455X (gfx1250) — matrix operand order repaired and checked by reading, not compiled, not run on hardware
//
#include <hip/hip_runtime.h>

typedef __attribute__((ext_vector_type(16))) _Float16 v16h;
typedef __attribute__((ext_vector_type(8)))  float    v8f;
typedef __attribute__((ext_vector_type(4)))  unsigned int v4u;
typedef __attribute__((ext_vector_type(8)))  int      v8i;
typedef __attribute__((ext_vector_type(4)))  int      v4i;

#define TOKS 2048
#define DIM  1024
#define NEXP 8
#define IDIM 1024
#define DM1  1023
#define IM1  1023
#define MAX_RTILES 136            // worst-case sum_e ceil(cnt_e/16) <= 135
#define SH_TILES   (TOKS / 16)    // 128 shared-expert tiles
#define KEPS 1e-8f
#define WPAD 1048576ull           // padded [1024,1024] f16 weight matrix elems

#if defined(__has_builtin)
#if __has_builtin(__builtin_amdgcn_tensor_load_to_lds) && \
    __has_builtin(__builtin_amdgcn_s_wait_tensorcnt)
#define HAVE_TDM 1
#endif
#endif
#ifndef HAVE_TDM
#define HAVE_TDM 0
#endif

// ---------------------------------------------------------------------------
// WMMA helpers (gfx1250, wave32): D(16x16,f32) = A(16x32,f16) x B(32x16,f16) + C
// A layout  : lane L holds row M=L%16; K = base16 + j,                 base16 = (L<16)?0:16
// B layout  : lane L holds col N=L%16; K = base16 + j,                 base16 = (L<16)?0:16
// C/D layout: VGPR r -> M = r + (L<16?0:8), N = L%16
// ---------------------------------------------------------------------------
__device__ __forceinline__ v8f wmma16(v16h a, v16h b, v8f c) {
  return __builtin_amdgcn_wmma_f32_16x16x32_f16(false, a, false, b, (short)0, c,
                                                false, false);
}

__device__ __forceinline__ v16h load_a_lds(const _Float16* lds_row, int k0) {
  // 16 contiguous halves at k0 (16B aligned)
  union { v16h v; uint4 q[2]; } u;
  u.q[0] = *(const uint4*)(lds_row + k0);
  u.q[1] = *(const uint4*)(lds_row + k0 + 8);
  return u.v;
}

// f16 pre-converted weights (row 1023 zero-padded): 16 contiguous halves.
__device__ __forceinline__ v16h load_b_f16(const _Float16* __restrict__ W,
                                           int n, int k) {
  union { v16h v; uint4 q[2]; } u;
  const _Float16* p = W + ((size_t)n << 10) + k;
  u.q[0] = *(const uint4*)(p);
  u.q[1] = *(const uint4*)(p + 8);
  return u.v;
}

// f32 fallback: branch-free row clamp; clamped lane feeds a discarded column.
__device__ __forceinline__ v16h load_b_f32(const float* __restrict__ W, int n,
                                           int k, int nlimit) {
  int nc = (n < nlimit) ? n : (nlimit - 1);
  const float4* p = (const float4*)(W + (size_t)nc * DIM + k);  // 64B aligned
  float4 f0 = p[0], f1 = p[1], f2 = p[2], f3 = p[3];
  v16h b;
  b[0]  = (_Float16)f0.x; b[1]  = (_Float16)f0.y;
  b[2]  = (_Float16)f0.z; b[3]  = (_Float16)f0.w;
  b[4]  = (_Float16)f1.x; b[5]  = (_Float16)f1.y;
  b[6]  = (_Float16)f1.z; b[7]  = (_Float16)f1.w;
  b[8]  = (_Float16)f2.x; b[9]  = (_Float16)f2.y;
  b[10] = (_Float16)f2.z; b[11] = (_Float16)f2.w;
  b[12] = (_Float16)f3.x; b[13] = (_Float16)f3.y;
  b[14] = (_Float16)f3.z; b[15] = (_Float16)f3.w;
  return b;
}

// ---------------------------------------------------------------------------
// 0) Weight precompression: [1023,1024] f32 -> [1024,1024] f16, row 1023 = 0.
//    grid = (1024, nmat); 4 elements/thread; 1047552 divisible by 4.
// ---------------------------------------------------------------------------
__global__ __launch_bounds__(256) void lmoe_convw_kernel(
    const float* __restrict__ src, _Float16* __restrict__ dst) {
  const size_t m = blockIdx.y;
  const int i = (blockIdx.x * 256 + threadIdx.x) * 4;
  union { _Float16 h[4]; uint2 u; } o;
  if (i < DM1 * DIM) {
    const float4 v = *(const float4*)(src + m * ((size_t)DM1 * DIM) + i);
    o.h[0] = (_Float16)v.x; o.h[1] = (_Float16)v.y;
    o.h[2] = (_Float16)v.z; o.h[3] = (_Float16)v.w;
  } else {
    o.h[0] = (_Float16)0.f; o.h[1] = (_Float16)0.f;
    o.h[2] = (_Float16)0.f; o.h[3] = (_Float16)0.f;
  }
  *(uint2*)(dst + m * WPAD + i) = o.u;
}

// ---------------------------------------------------------------------------
// 1) Gating: logits = x[:,1:] @ gate_w.T ; softmax ; argmax(score+bias) -> top-1
// ---------------------------------------------------------------------------
__global__ __launch_bounds__(256) void lmoe_gate_kernel(
    const float* __restrict__ x, const float* __restrict__ gw,
    const float* __restrict__ gb, int* __restrict__ tokExp,
    float* __restrict__ tokW) {
  __shared__ float red[NEXP][256];
  const int t = blockIdx.x, tid = threadIdx.x;
  float acc[NEXP];
#pragma unroll
  for (int e = 0; e < NEXP; e++) acc[e] = 0.f;
  for (int d = tid; d < DM1; d += 256) {
    float xv = x[(size_t)t * DIM + 1 + d];
#pragma unroll
    for (int e = 0; e < NEXP; e++) acc[e] += xv * gw[e * DM1 + d];
  }
#pragma unroll
  for (int e = 0; e < NEXP; e++) red[e][tid] = acc[e];
  __syncthreads();
  for (int s = 128; s > 0; s >>= 1) {
    if (tid < s) {
#pragma unroll
      for (int e = 0; e < NEXP; e++) red[e][tid] += red[e][tid + s];
    }
    __syncthreads();
  }
  if (tid == 0) {
    float m = -3.4e38f;
    for (int e = 0; e < NEXP; e++) m = fmaxf(m, red[e][0]);
    float p[NEXP], s = 0.f;
    for (int e = 0; e < NEXP; e++) { p[e] = __expf(red[e][0] - m); s += p[e]; }
    float inv = __fdividef(1.f, s);
    int best = 0; float bb = -3.4e38f, bwt = 0.f;
    for (int e = 0; e < NEXP; e++) {
      float sc = p[e] * inv;
      float bi = sc + gb[e];
      if (bi > bb) { bb = bi; best = e; bwt = sc; }
    }
    tokExp[t] = best;
    tokW[t] = bwt;
  }
}

// ---------------------------------------------------------------------------
// 2) Bucket tokens by expert into 16-token tiles (pad with -1)
// ---------------------------------------------------------------------------
__global__ __launch_bounds__(256) void lmoe_permute_kernel(
    const int* __restrict__ tokExp, int* __restrict__ order,
    int* __restrict__ tileExpert) {
  __shared__ int cnt[NEXP], segB[NEXP], cur[NEXP];
  const int tid = threadIdx.x;
  if (tid < NEXP) { cnt[tid] = 0; cur[tid] = 0; }
  for (int i = tid; i < MAX_RTILES * 16; i += 256) order[i] = -1;
  if (tid < MAX_RTILES) tileExpert[tid] = -1;
  __syncthreads();
  for (int t = tid; t < TOKS; t += 256) atomicAdd(&cnt[tokExp[t]], 1);
  __syncthreads();
  if (tid == 0) {
    int base = 0, tile = 0;
    for (int e = 0; e < NEXP; e++) {
      segB[e] = base;
      int nt = (cnt[e] + 15) >> 4;
      for (int k = 0; k < nt; k++) tileExpert[tile++] = e;
      base += nt << 4;
    }
  }
  __syncthreads();
  for (int t = tid; t < TOKS; t += 256) {
    int e = tokExp[t];
    int p = atomicAdd(&cur[e], 1);
    order[segB[e] + p] = t;
  }
}

// ---------------------------------------------------------------------------
// 3) Grouped Lorentz-SwiGLU FFN over 16-token single-expert tiles.
//    F16W=true : B operands from pre-converted padded f16 weights in ws.
//    F16W=false: B operands converted from f32 on the fly (small-ws fallback).
// ---------------------------------------------------------------------------
#define STAGE_ROWS 4  // TDM gather chunk (rows of x staged per descriptor)

template <bool F16W>
__global__ __launch_bounds__(256) void lmoe_ffn_kernel(
    const float* __restrict__ x, const float* __restrict__ W1f,
    const float* __restrict__ W3f, const float* __restrict__ W2f,
    const float* __restrict__ Ws1f, const float* __restrict__ Ws3f,
    const float* __restrict__ Ws2f, const _Float16* __restrict__ W1h,
    const _Float16* __restrict__ W3h, const _Float16* __restrict__ W2h,
    const _Float16* __restrict__ Wsh,  // [Ws1h | Ws3h | Ws2h] packed
    const float* __restrict__ tokW, const int* __restrict__ order,
    const int* __restrict__ tileExpert, float* __restrict__ rbuf,
    float* __restrict__ zbuf) {
  __shared__ _Float16 tileLDS[16 * DIM];        // x-tile, then h-tile (32 KB)
  __shared__ float    stage[STAGE_ROWS * DIM];  // TDM f32 landing pad (16 KB)
  __shared__ float rowred[16];
  __shared__ int   stoks[16];
  __shared__ float sw[16];
  __shared__ int   sExp;

  const int tid = threadIdx.x;
  const int tile = blockIdx.x;

  if (tid == 0) sExp = (tile >= MAX_RTILES) ? NEXP : tileExpert[tile];
  __syncthreads();
  const int e = sExp;
  if (e < 0) return;  // unused padded routed tile
  const bool isShared = (e == NEXP);

  if (tid < 16) {
    int t = isShared ? ((tile - MAX_RTILES) * 16 + tid) : order[tile * 16 + tid];
    stoks[tid] = t;
    sw[tid] = (!isShared && t >= 0) ? tokW[t] : 1.0f;
    rowred[tid] = 0.f;
  }
  __syncthreads();

  const float* W1p32 = isShared ? Ws1f : (W1f + (size_t)e * IM1 * DIM);
  const float* W3p32 = isShared ? Ws3f : (W3f + (size_t)e * IM1 * DIM);
  const float* W2p32 = isShared ? Ws2f : (W2f + (size_t)e * DM1 * IDIM);
  const _Float16* W1p16 = isShared ? (Wsh + 0 * WPAD) : (W1h + (size_t)e * WPAD);
  const _Float16* W3p16 = isShared ? (Wsh + 1 * WPAD) : (W3h + (size_t)e * WPAD);
  const _Float16* W2p16 = isShared ? (Wsh + 2 * WPAD) : (W2h + (size_t)e * WPAD);
  float* obuf = isShared ? zbuf : rbuf;

  const int wave = tid >> 5;

#if HAVE_TDM
  // ---- stage x tile via Tensor Data Mover gather (D# per cdna5 08_async) --
  // x is a 2D tensor [2048,1024] f32; row indices = token ids (u16). Padded
  // tokens (-1 -> 0xFFFF) land OOB and read back zeros per the D# OOB rule.
  {
    const unsigned long long gaddr = (unsigned long long)(const void*)x;
    const unsigned ldsOff = (unsigned)(size_t)(&stage[0]);  // addr[31:0] = LDS offset
    v4u g0;
    g0[0] = 1u | (1u << 31);  // count=1, is_restore=0, 16b indices, gather_mode=1
    g0[1] = ldsOff;           // lds_addr
    g0[2] = (unsigned)(gaddr & 0xFFFFFFFFull);            // global_addr[31:0]
    g0[3] = (unsigned)((gaddr >> 32) & 0x1FFFFFFull) | (2u << 30);  // [56:32], type=2
    v8i g1;
    g1[0] = (int)(2u << 16);        // workgroup_mask=0, data_size=4B
    g1[1] = (int)(1024u << 16);     // tensor_dim0 = 1024 (low bits at [63:48])
    g1[2] = (int)(2048u << 16);     // tensor_dim1 = 2048 (low bits at [95:80])
    g1[3] = (int)(1024u << 16);     // tile_dim0 = 1024
    g1[4] = STAGE_ROWS;             // tile_dim1 = #gather indices
    g1[5] = 1024;                   // tensor_dim0_stride
    g1[6] = 0;
    g1[7] = 0;
    for (int c = 0; c < 16 / STAGE_ROWS; c++) {
      unsigned i0 = (unsigned short)stoks[c * STAGE_ROWS + 0];
      unsigned i1 = (unsigned short)stoks[c * STAGE_ROWS + 1];
      unsigned i2 = (unsigned short)stoks[c * STAGE_ROWS + 2];
      unsigned i3 = (unsigned short)stoks[c * STAGE_ROWS + 3];
      v4i g2;
      g2[0] = (int)(i0 | (i1 << 16));
      g2[1] = (int)(i2 | (i3 << 16));
      g2[2] = 0; g2[3] = 0;
      v4i g3; g3[0] = 0; g3[1] = 0; g3[2] = 0; g3[3] = 0;
      if (wave == 0) {
#if __clang_major__ >= 23
        v8i gx; gx[0]=0; gx[1]=0; gx[2]=0; gx[3]=0; gx[4]=0; gx[5]=0; gx[6]=0; gx[7]=0;
        __builtin_amdgcn_tensor_load_to_lds(g0, g1, g2, g3, gx, 0);
#else
        __builtin_amdgcn_tensor_load_to_lds(g0, g1, g2, g3, 0);
#endif
        __builtin_amdgcn_s_wait_tensorcnt(0);
      }
      __syncthreads();  // TDM data visible in LDS; convert f32 -> f16
      for (int i = tid; i < STAGE_ROWS * DIM; i += 256)
        tileLDS[c * STAGE_ROWS * DIM + i] = (_Float16)stage[i];
      __syncthreads();  // staging buffer free for next chunk
    }
  }
#else
  // fallback: direct gather (prescale sqrt(c_e/c) == 1, plain f32->f16)
  for (int i = tid; i < 16 * DIM; i += 256) {
    int row = i >> 10, col = i & (DIM - 1);
    int t = stoks[row];
    tileLDS[i] = (t >= 0) ? (_Float16)x[(size_t)t * DIM + col] : (_Float16)0.f;
  }
  __syncthreads();
#endif

  const int lane = tid & 31;
  const int mrow = lane & 15;
  const bool lo = lane < 16;
  const int base8  = lo ? 0 : 8;
  const int base16 = lo ? 0 : 16;
  const int mbase  = lo ? 0 : 8;
  const int nwave  = wave * 128;

  // ---- GEMM1: s1 = x @ W1^T, s3 = x @ W3^T (per-wave 16x128 slab) ----
  v8f acc1[8], acc3[8];
#pragma unroll
  for (int i = 0; i < 8; i++) {
#pragma unroll
    for (int r = 0; r < 8; r++) { acc1[i][r] = 0.f; acc3[i][r] = 0.f; }
  }
  for (int kb = 0; kb < DIM; kb += 32) {
    const int kB = kb + base16;
    v16h a = load_a_lds(&tileLDS[mrow * DIM], kB);
    // software pipeline: prefetch fragment nt+1 before issuing wmma nt
    v16h b1 = F16W ? load_b_f16(W1p16, nwave + mrow, kB)
                   : load_b_f32(W1p32, nwave + mrow, kB, IM1);
    v16h b3 = F16W ? load_b_f16(W3p16, nwave + mrow, kB)
                   : load_b_f32(W3p32, nwave + mrow, kB, IM1);
#pragma unroll
    for (int nt = 0; nt < 8; nt++) {
      v16h nb1, nb3;
      if (nt < 7) {
        int n = nwave + (nt + 1) * 16 + mrow;
        nb1 = F16W ? load_b_f16(W1p16, n, kB) : load_b_f32(W1p32, n, kB, IM1);
        nb3 = F16W ? load_b_f16(W3p16, n, kB) : load_b_f32(W3p32, n, kB, IM1);
      }
      acc1[nt] = wmma16(a, b1, acc1[nt]);
      acc3[nt] = wmma16(a, b3, acc3[nt]);
      if (nt < 7) { b1 = nb1; b3 = nb3; }
    }
  }
  __syncthreads();  // x-tile reads complete; tileLDS becomes h-tile

  // ---- sp = silu(s1)*s3 ; Lorentz time lift t = sqrt(|sp|^2 + c_mid) ----
  {
    float rp[8];
#pragma unroll
    for (int r = 0; r < 8; r++) rp[r] = 0.f;
#pragma unroll
    for (int nt = 0; nt < 8; nt++) {
      int n = nwave + nt * 16 + mrow;
#pragma unroll
      for (int r = 0; r < 8; r++) {
        float v1 = acc1[nt][r], v3 = acc3[nt][r];
        float sp = __fdividef(v1, 1.f + __expf(-v1)) * v3;
        if (n < IM1) {
          tileLDS[(mbase + r) * DIM + 1 + n] = (_Float16)sp;
          rp[r] += sp * sp;
        }
      }
    }
#pragma unroll
    for (int r = 0; r < 8; r++) atomicAdd(&rowred[mbase + r], rp[r]);
  }
  __syncthreads();
  if (tid < 16) {
    float v = rowred[tid];
    rowred[tid] = 0.f;  // reused for the output time lift
    tileLDS[tid * DIM] = (_Float16)sqrtf(fmaxf(v + 1.0f, KEPS));  // c_mid = 1
  }
  __syncthreads();

  // ---- GEMM2: o = h @ W2^T ----
  v8f acc2[8];
#pragma unroll
  for (int i = 0; i < 8; i++) {
#pragma unroll
    for (int r = 0; r < 8; r++) acc2[i][r] = 0.f;
  }
  for (int kb = 0; kb < IDIM; kb += 32) {
    const int kB = kb + base16;
    v16h a = load_a_lds(&tileLDS[mrow * DIM], kB);
    v16h b2 = F16W ? load_b_f16(W2p16, nwave + mrow, kB)
                   : load_b_f32(W2p32, nwave + mrow, kB, DM1);
#pragma unroll
    for (int nt = 0; nt < 8; nt++) {
      v16h nb2;
      if (nt < 7) {
        int n = nwave + (nt + 1) * 16 + mrow;
        nb2 = F16W ? load_b_f16(W2p16, n, kB) : load_b_f32(W2p32, n, kB, DM1);
      }
      acc2[nt] = wmma16(a, b2, acc2[nt]);
      if (nt < 7) b2 = nb2;
    }
  }

  // ---- output time lift + scaled store (routed: *gate_w ; shared: *1) ----
  {
    float rp[8];
#pragma unroll
    for (int r = 0; r < 8; r++) rp[r] = 0.f;
#pragma unroll
    for (int nt = 0; nt < 8; nt++) {
      int n = nwave + nt * 16 + mrow;
      if (n < DM1) {
#pragma unroll
        for (int r = 0; r < 8; r++) { float o = acc2[nt][r]; rp[r] += o * o; }
      }
    }
#pragma unroll
    for (int r = 0; r < 8; r++) atomicAdd(&rowred[mbase + r], rp[r]);
  }
  __syncthreads();
#pragma unroll
  for (int nt = 0; nt < 8; nt++) {
    int n = nwave + nt * 16 + mrow;
    if (n < DM1) {
#pragma unroll
      for (int r = 0; r < 8; r++) {
        int row = mbase + r;
        int t = stoks[row];
        if (t >= 0) obuf[(size_t)t * DIM + 1 + n] = sw[row] * acc2[nt][r];
      }
    }
  }
  if (wave == 0 && mrow == 0) {  // one writer per row for the time component
#pragma unroll
    for (int r = 0; r < 8; r++) {
      int row = mbase + r;
      int t = stoks[row];
      if (t >= 0)
        obuf[(size_t)t * DIM] = sw[row] * sqrtf(fmaxf(rowred[row] + 1.0f, KEPS));
    }
  }
}

// ---------------------------------------------------------------------------
// 4) LResNet combine + Lorentz renormalization
//    comb = z + 2*(y0 + r);  out = comb / sqrt(max(|<comb,comb>_L|, eps))
// ---------------------------------------------------------------------------
__global__ __launch_bounds__(256) void lmoe_combine_kernel(
    const float* __restrict__ rbuf, const float* __restrict__ zbuf,
    float* __restrict__ out) {
  __shared__ float red[256];
  __shared__ float nrmS;
  const int t = blockIdx.x, tid = threadIdx.x;
  float part = 0.f;
  for (int d = tid; d < DIM; d += 256) {
    float c = zbuf[(size_t)t * DIM + d] + 2.f * rbuf[(size_t)t * DIM + d] +
              ((d == 0) ? 2.f : 0.f);  // y0 = [sqrt(C)=1, 0...], scale 2
    part += (d == 0) ? -(c * c) : (c * c);
  }
  red[tid] = part;
  __syncthreads();
  for (int s = 128; s > 0; s >>= 1) {
    if (tid < s) red[tid] += red[tid + s];
    __syncthreads();
  }
  if (tid == 0) nrmS = 1.f / sqrtf(fmaxf(fabsf(red[0]), KEPS));
  __syncthreads();
  float nv = nrmS;
  for (int d = tid; d < DIM; d += 256) {
    float c = zbuf[(size_t)t * DIM + d] + 2.f * rbuf[(size_t)t * DIM + d] +
              ((d == 0) ? 2.f : 0.f);
    out[(size_t)t * DIM + d] = c * nv;
  }
}

// ---------------------------------------------------------------------------
extern "C" void kernel_launch(void* const* d_in, const int* in_sizes, int n_in,
                              void* d_out, int out_size, void* d_ws,
                              size_t ws_size, hipStream_t stream) {
  const float* x      = (const float*)d_in[0];
  const float* gate_w = (const float*)d_in[1];
  const float* gate_b = (const float*)d_in[2];
  const float* W1     = (const float*)d_in[3];
  const float* W3     = (const float*)d_in[4];
  const float* W2     = (const float*)d_in[5];
  const float* Ws1    = (const float*)d_in[6];
  const float* Ws3    = (const float*)d_in[7];
  const float* Ws2    = (const float*)d_in[8];
  float* out = (float*)d_out;

  char* ws = (char*)d_ws;
  const size_t MB8 = (size_t)TOKS * DIM * 4;  // 8 MB
  int*   tokExp     = (int*)(ws + 0);         //  8 KB
  float* tokW       = (float*)(ws + 8192);    //  8 KB
  int*   order      = (int*)(ws + 16384);     //  ~8.5 KB (reserve 12 KB)
  int*   tileExpert = (int*)(ws + 28672);     //  <4 KB
  float* rbuf       = (float*)(ws + 32768);   //  8 MB routed contribution
  float* zbuf       = (float*)(ws + 32768 + MB8);  // 8 MB shared-expert out
  // optional f16 weight cache (padded [1024,1024], row 1023 zero)
  char* wh   = ws + 32768 + 2 * MB8;
  _Float16* W1h = (_Float16*)(wh);
  _Float16* W3h = (_Float16*)(wh + 1 * NEXP * WPAD * 2);
  _Float16* W2h = (_Float16*)(wh + 2 * NEXP * WPAD * 2);
  _Float16* Wsh = (_Float16*)(wh + 3 * NEXP * WPAD * 2);
  const size_t need_f16 =
      32768 + 2 * MB8 + (3 * NEXP + 3) * WPAD * 2;  // ~70 MB

  lmoe_gate_kernel<<<TOKS, 256, 0, stream>>>(x, gate_w, gate_b, tokExp, tokW);
  lmoe_permute_kernel<<<1, 256, 0, stream>>>(tokExp, order, tileExpert);

  if (ws_size >= need_f16) {
    lmoe_convw_kernel<<<dim3(1024, NEXP), 256, 0, stream>>>(W1, W1h);
    lmoe_convw_kernel<<<dim3(1024, NEXP), 256, 0, stream>>>(W3, W3h);
    lmoe_convw_kernel<<<dim3(1024, NEXP), 256, 0, stream>>>(W2, W2h);
    lmoe_convw_kernel<<<dim3(1024, 1), 256, 0, stream>>>(Ws1, Wsh + 0 * WPAD);
    lmoe_convw_kernel<<<dim3(1024, 1), 256, 0, stream>>>(Ws3, Wsh + 1 * WPAD);
    lmoe_convw_kernel<<<dim3(1024, 1), 256, 0, stream>>>(Ws2, Wsh + 2 * WPAD);
    lmoe_ffn_kernel<true><<<MAX_RTILES + SH_TILES, 256, 0, stream>>>(
        x, W1, W3, W2, Ws1, Ws3, Ws2, W1h, W3h, W2h, Wsh, tokW, order,
        tileExpert, rbuf, zbuf);
  } else {
    lmoe_ffn_kernel<false><<<MAX_RTILES + SH_TILES, 256, 0, stream>>>(
        x, W1, W3, W2, Ws1, Ws3, Ws2, W1h, W3h, W2h, Wsh, tokW, order,
        tileExpert, rbuf, zbuf);
  }
  lmoe_combine_kernel<<<TOKS, 256, 0, stream>>>(rbuf, zbuf, out);
}
